// DeepSeekMoE_41936060678344
// MI455X (gfx1250) — matrix operand order repaired and checked by reading, not compiled, not run on hardware
//
#include <hip/hip_runtime.h>
#include <hip/hip_bf16.h>
#include <cstdint>

// ---------------------------------------------------------------------------
// MI455X (gfx1250) DeepSeek MoE: bf16 WMMA GEMMs, f32 accumulate.
// AI ~1000 flop/B -> compute-bound -> v_wmma_f32_16x16x32_bf16 (K=32/issue).
// Weights are converted fp32->bf16 *transposed* (Wt[N,K]) so both LDS staging
// and per-lane B-fragment assembly are contiguous 128-bit accesses.
// ---------------------------------------------------------------------------

typedef __attribute__((ext_vector_type(16))) __bf16        v16bf;
typedef __attribute__((ext_vector_type(8)))  float         v8f;
typedef __attribute__((ext_vector_type(4)))  unsigned int  v4u;

union FragBF { v16bf bf; v4u u[2]; };  // 32B: one WMMA A/B operand (8 VGPRs)

#define NEXP 8

#define MT 128   // output tile rows per block
#define NT 128   // output tile cols per block
#define KT 32    // K step (bf16 WMMA K)

// ---------------- fp32 -> bf16 streaming convert (no transpose: X) ----------
__global__ void cvt_f32_to_bf16(const float* __restrict__ in,
                                __bf16* __restrict__ out, int n) {
    int i = blockIdx.x * blockDim.x + threadIdx.x;
    if (i < n) out[i] = (__bf16)in[i];
}

// ---------------- fp32 [K,N] -> bf16 [N,K] tiled transpose-convert ----------
__global__ __launch_bounds__(256) void cvt_T_f32_to_bf16(
    const float* __restrict__ in,  // [K, N] fp32 row-major
    __bf16* __restrict__ out,      // [N, K] bf16 row-major
    int K, int N) {
    __shared__ float tile[32][33];  // +1 pad: conflict-free transpose
    const int kb = blockIdx.x % (K / 32);
    const int nb = blockIdx.x / (K / 32);
    const int k0 = kb * 32, n0 = nb * 32;
    const int tx = threadIdx.x & 31;
    const int ty = threadIdx.x >> 5;  // 0..7
#pragma unroll
    for (int r = ty; r < 32; r += 8)
        tile[r][tx] = in[(size_t)(k0 + r) * N + n0 + tx];
    __syncthreads();
#pragma unroll
    for (int r = ty; r < 32; r += 8)
        out[(size_t)(n0 + r) * K + k0 + tx] = (__bf16)tile[tx][r];
}

__global__ void zero_f32(float* __restrict__ p, int n) {
    int i = blockIdx.x * blockDim.x + threadIdx.x;
    if (i < n) p[i] = 0.f;
}

// ---------------- gate: logits -> softmax -> top-2 scatter ----------------
__global__ void gate_topk_kernel(const float* __restrict__ X,
                                 const float* __restrict__ gW,
                                 float* __restrict__ gates, int T, int D) {
    const int t = blockIdx.x * blockDim.x + threadIdx.x;
    if (t >= T) return;
    float acc[NEXP];
#pragma unroll
    for (int e = 0; e < NEXP; ++e) acc[e] = 0.f;
    const float* x = X + (size_t)t * D;
    for (int d = 0; d < D; ++d) {
        const float xv = x[d];
#pragma unroll
        for (int e = 0; e < NEXP; ++e) acc[e] += xv * gW[e * D + d];  // gW uniform -> s_load
    }
    float m = acc[0];
#pragma unroll
    for (int e = 1; e < NEXP; ++e) m = fmaxf(m, acc[e]);
    float p[NEXP], sum = 0.f;
#pragma unroll
    for (int e = 0; e < NEXP; ++e) { p[e] = __expf(acc[e] - m); sum += p[e]; }
    const float inv = 1.f / sum;
#pragma unroll
    for (int e = 0; e < NEXP; ++e) p[e] *= inv;
    int i1 = 0;
#pragma unroll
    for (int e = 1; e < NEXP; ++e) if (p[e] > p[i1]) i1 = e;
    int i2 = (i1 == 0) ? 1 : 0;
#pragma unroll
    for (int e = 0; e < NEXP; ++e) if (e != i1 && p[e] > p[i2]) i2 = e;
#pragma unroll
    for (int e = 0; e < NEXP; ++e)
        gates[(size_t)t * NEXP + e] = (e == i1 || e == i2) ? p[e] : 0.f;
}

// ---------------- fused GEMM1: H = silu(X*W1) .* (X*W3) ---------------------
// X:[T,K] bf16 rm, W1t/W3t:[N,K] bf16 rm (pre-transposed), H:[T,N] bf16.
// 128x128 block tile, 8 waves each own 32x64 (2 A-frags x 4 B-frags).
__global__ __launch_bounds__(256) void moe_gemm1_silu_mul(
    const __bf16* __restrict__ X, const __bf16* __restrict__ W1t,
    const __bf16* __restrict__ W3t, __bf16* __restrict__ H,
    int Kdim, int N) {
    __shared__ __align__(16) __bf16 Xs[MT * KT];    // 8 KB, [m][k]
    __shared__ __align__(16) __bf16 W1s[NT * KT];   // 8 KB, [n][k]
    __shared__ __align__(16) __bf16 W3s[NT * KT];   // 8 KB, [n][k]

    const int nb = N / NT;
    const int m0 = (blockIdx.x / nb) * MT;
    const int n0 = (blockIdx.x % nb) * NT;

    const int tid  = threadIdx.x;
    const int lane = tid & 31;
    const int wave = tid >> 5;
    const int wm   = (wave >> 1) * 32;  // wave row offset in tile (0..96)
    const int wn   = (wave & 1) * 64;   // wave col offset in tile (0/64)
    const int mrow = lane & 15;
    const int hlf  = lane >> 4;

    const v8f vzero = {0.f, 0.f, 0.f, 0.f, 0.f, 0.f, 0.f, 0.f};
    v8f acc1[2][4], acc3[2][4];
#pragma unroll
    for (int mi = 0; mi < 2; ++mi)
#pragma unroll
        for (int j = 0; j < 4; ++j) { acc1[mi][j] = vzero; acc3[mi][j] = vzero; }

    for (int k0 = 0; k0 < Kdim; k0 += KT) {
        // cooperative 128-bit staging: 512 chunks of 8 bf16 per array
#pragma unroll
        for (int ch = 0; ch < 2; ++ch) {
            const int id = tid + ch * 256;
            const int r  = id >> 2;
            const int c  = (id & 3) << 3;
            *reinterpret_cast<v4u*>(&Xs[r * KT + c]) =
                *reinterpret_cast<const v4u*>(&X[(size_t)(m0 + r) * Kdim + k0 + c]);
            *reinterpret_cast<v4u*>(&W1s[r * KT + c]) =
                *reinterpret_cast<const v4u*>(&W1t[(size_t)(n0 + r) * Kdim + k0 + c]);
            *reinterpret_cast<v4u*>(&W3s[r * KT + c]) =
                *reinterpret_cast<const v4u*>(&W3t[(size_t)(n0 + r) * Kdim + k0 + c]);
        }
        __syncthreads();

        // A fragments: lane = row m, K = hlf*16 .. +15 contiguous (matches B)
        FragBF a[2];
#pragma unroll
        for (int mi = 0; mi < 2; ++mi) {
            const __bf16* xr = &Xs[(wm + mi * 16 + mrow) * KT];
            a[mi].u[0] = *reinterpret_cast<const v4u*>(xr + hlf * 16);
            a[mi].u[1] = *reinterpret_cast<const v4u*>(xr + hlf * 16 + 8);
        }
        // B fragments: lane = col n, K = hlf*16 .. +15 contiguous in Wt row
#pragma unroll
        for (int j = 0; j < 4; ++j) {
            const __bf16* w1r = &W1s[(wn + j * 16 + mrow) * KT + hlf * 16];
            const __bf16* w3r = &W3s[(wn + j * 16 + mrow) * KT + hlf * 16];
            FragBF b1, b3;
            b1.u[0] = *reinterpret_cast<const v4u*>(w1r);
            b1.u[1] = *reinterpret_cast<const v4u*>(w1r + 8);
            b3.u[0] = *reinterpret_cast<const v4u*>(w3r);
            b3.u[1] = *reinterpret_cast<const v4u*>(w3r + 8);
#pragma unroll
            for (int mi = 0; mi < 2; ++mi) {
                acc1[mi][j] = __builtin_amdgcn_wmma_f32_16x16x32_bf16(
                    false, a[mi].bf, false, b1.bf, (short)0, acc1[mi][j], false, false);
                acc3[mi][j] = __builtin_amdgcn_wmma_f32_16x16x32_bf16(
                    false, a[mi].bf, false, b3.bf, (short)0, acc3[mi][j], false, false);
            }
        }
        __syncthreads();
    }

    // epilogue: h = silu(g) * u (C layout: row = hlf*8+i, col = lane%16)
#pragma unroll
    for (int mi = 0; mi < 2; ++mi)
#pragma unroll
        for (int j = 0; j < 4; ++j) {
            const int col = n0 + wn + j * 16 + mrow;
#pragma unroll
            for (int i = 0; i < 8; ++i) {
                const int row = m0 + wm + mi * 16 + hlf * 8 + i;
                const float g = acc1[mi][j][i];
                const float u = acc3[mi][j][i];
                const float s = g / (1.0f + __expf(-g));
                H[(size_t)row * N + col] = (__bf16)(s * u);
            }
        }
}

// ---------------- GEMM2: OUT += scale(t) * (A * W2) -------------------------
// A:[T,K] bf16 rm, W2t:[N,K] bf16 rm (pre-transposed), OUT:[T,N] f32.
// scale(t) = alpha * gates[t,eidx] (routed) or alpha (shared, gates==nullptr).
__global__ __launch_bounds__(256) void moe_gemm2_acc(
    const __bf16* __restrict__ A, const __bf16* __restrict__ W2t,
    float* __restrict__ OUT, const float* __restrict__ gates,
    int eidx, float alpha, int Kdim, int N) {
    __shared__ __align__(16) __bf16 As[MT * KT];  // 8 KB
    __shared__ __align__(16) __bf16 Ws[NT * KT];  // 8 KB

    const int nb = N / NT;
    const int m0 = (blockIdx.x / nb) * MT;
    const int n0 = (blockIdx.x % nb) * NT;

    const int tid  = threadIdx.x;
    const int lane = tid & 31;
    const int wave = tid >> 5;
    const int wm   = (wave >> 1) * 32;
    const int wn   = (wave & 1) * 64;
    const int mrow = lane & 15;
    const int hlf  = lane >> 4;

    const v8f vzero = {0.f, 0.f, 0.f, 0.f, 0.f, 0.f, 0.f, 0.f};
    v8f acc[2][4];
#pragma unroll
    for (int mi = 0; mi < 2; ++mi)
#pragma unroll
        for (int j = 0; j < 4; ++j) acc[mi][j] = vzero;

    for (int k0 = 0; k0 < Kdim; k0 += KT) {
#pragma unroll
        for (int ch = 0; ch < 2; ++ch) {
            const int id = tid + ch * 256;
            const int r  = id >> 2;
            const int c  = (id & 3) << 3;
            *reinterpret_cast<v4u*>(&As[r * KT + c]) =
                *reinterpret_cast<const v4u*>(&A[(size_t)(m0 + r) * Kdim + k0 + c]);
            *reinterpret_cast<v4u*>(&Ws[r * KT + c]) =
                *reinterpret_cast<const v4u*>(&W2t[(size_t)(n0 + r) * Kdim + k0 + c]);
        }
        __syncthreads();

        FragBF a[2];
#pragma unroll
        for (int mi = 0; mi < 2; ++mi) {
            const __bf16* ar = &As[(wm + mi * 16 + mrow) * KT];
            a[mi].u[0] = *reinterpret_cast<const v4u*>(ar + hlf * 16);
            a[mi].u[1] = *reinterpret_cast<const v4u*>(ar + hlf * 16 + 8);
        }
#pragma unroll
        for (int j = 0; j < 4; ++j) {
            const __bf16* wr = &Ws[(wn + j * 16 + mrow) * KT + hlf * 16];
            FragBF b;
            b.u[0] = *reinterpret_cast<const v4u*>(wr);
            b.u[1] = *reinterpret_cast<const v4u*>(wr + 8);
#pragma unroll
            for (int mi = 0; mi < 2; ++mi)
                acc[mi][j] = __builtin_amdgcn_wmma_f32_16x16x32_bf16(
                    false, a[mi].bf, false, b.bf, (short)0, acc[mi][j], false, false);
        }
        __syncthreads();
    }

#pragma unroll
    for (int mi = 0; mi < 2; ++mi)
#pragma unroll
        for (int j = 0; j < 4; ++j) {
            const int col = n0 + wn + j * 16 + mrow;
#pragma unroll
            for (int i = 0; i < 8; ++i) {
                const int row = m0 + wm + mi * 16 + hlf * 8 + i;
                float c = alpha;
                if (gates) c *= gates[(size_t)row * NEXP + eidx];
                OUT[(size_t)row * N + col] += c * acc[mi][j][i];
            }
        }
}

// ---------------------------------------------------------------------------
extern "C" void kernel_launch(void* const* d_in, const int* in_sizes, int n_in,
                              void* d_out, int out_size, void* d_ws, size_t ws_size,
                              hipStream_t stream) {
    (void)in_sizes; (void)n_in; (void)out_size; (void)ws_size;
    const float* hs  = (const float*)d_in[0];  // [B,S,D] fp32
    const float* gW  = (const float*)d_in[1];  // [E,D]
    const float* w1e = (const float*)d_in[2];  // [E,D,F]
    const float* w3e = (const float*)d_in[3];  // [E,D,F]
    const float* w2e = (const float*)d_in[4];  // [E,F,D]
    const float* w1s = (const float*)d_in[5];  // [NS,D,FS]
    const float* w3s = (const float*)d_in[6];
    const float* w2s = (const float*)d_in[7];  // [NS,FS,D]
    float* out = (float*)d_out;

    const int T = 2048, D = 1024, F = 4096, FS = 512, E = 8, NS = 8;

    // workspace (~45 MB): per-expert bf16 staging; per-stage weights are L2-resident
    char* ws = (char*)d_ws;
    __bf16* Xb  = (__bf16*)(ws);                            // 4 MB  [T,D]
    __bf16* W1b = (__bf16*)(ws + (size_t)(4)  * (1 << 20)); // 8 MB  transposed [N,K]
    __bf16* W3b = (__bf16*)(ws + (size_t)(12) * (1 << 20)); // 8 MB
    __bf16* W2b = (__bf16*)(ws + (size_t)(20) * (1 << 20)); // 8 MB
    __bf16* Hb  = (__bf16*)(ws + (size_t)(28) * (1 << 20)); // 16 MB [T,F] max
    float*  gts = (float*)(ws + (size_t)(44) * (1 << 20));  // 64 KB [T,E]

    const int nX = T * D;
    cvt_f32_to_bf16<<<(nX + 255) / 256, 256, 0, stream>>>(hs, Xb, nX);
    gate_topk_kernel<<<(T + 127) / 128, 128, 0, stream>>>(hs, gW, gts, T, D);
    zero_f32<<<(T * D + 255) / 256, 256, 0, stream>>>(out, T * D);

    // routed experts: out += 0.5 * gate[t,e] * FFN_e(x)
    const int nW    = D * F;
    const int gT_DF = (D / 32) * (F / 32);  // transpose grid for [D,F] / [F,D]
    for (int e = 0; e < E; ++e) {
        cvt_T_f32_to_bf16<<<gT_DF, 256, 0, stream>>>(w1e + (size_t)e * nW, W1b, D, F);
        cvt_T_f32_to_bf16<<<gT_DF, 256, 0, stream>>>(w3e + (size_t)e * nW, W3b, D, F);
        moe_gemm1_silu_mul<<<(T / MT) * (F / NT), 256, 0, stream>>>(Xb, W1b, W3b, Hb, D, F);
        cvt_T_f32_to_bf16<<<gT_DF, 256, 0, stream>>>(w2e + (size_t)e * nW, W2b, F, D);
        moe_gemm2_acc<<<(T / MT) * (D / NT), 256, 0, stream>>>(Hb, W2b, out, gts, e, 0.5f, F, D);
    }

    // shared experts: out += (0.5 / NS) * FFN_n(x)
    const int nWs    = D * FS;
    const int gT_DFS = (D / 32) * (FS / 32);
    for (int n = 0; n < NS; ++n) {
        cvt_T_f32_to_bf16<<<gT_DFS, 256, 0, stream>>>(w1s + (size_t)n * nWs, W1b, D, FS);
        cvt_T_f32_to_bf16<<<gT_DFS, 256, 0, stream>>>(w3s + (size_t)n * nWs, W3b, D, FS);
        moe_gemm1_silu_mul<<<(T / MT) * (FS / NT), 256, 0, stream>>>(Xb, W1b, W3b, Hb, D, FS);
        cvt_T_f32_to_bf16<<<gT_DFS, 256, 0, stream>>>(w2s + (size_t)n * nWs, W2b, FS, D);
        moe_gemm2_acc<<<(T / MT) * (D / NT), 256, 0, stream>>>(Hb, W2b, out, nullptr, 0,
                                                               0.5f / (float)NS, FS, D);
    }
}
